// RealmRetriever_81819126988901
// MI455X (gfx1250) — matrix operand order repaired and checked by reading, not compiled, not run on hardware
//
#include <hip/hip_runtime.h>
#include <hip/hip_bf16.h>
#include <stdint.h>

#define NDOCS 500000
#define DIMS  768
#define NQ    64
#define TOPK  5
#define TILE_DOCS 128
#define NTILES ((NDOCS + TILE_DOCS - 1) / TILE_DOCS)   // 3907
#define KCHUNKS (DIMS / 32)                            // 24
#define QPITCH 776                                     // bf16 elems per q row in LDS (1552 B)

typedef __attribute__((ext_vector_type(16))) __bf16 v16bf;
typedef __attribute__((ext_vector_type(8)))  float  v8f;

static __device__ __forceinline__ bool better(float av, int ai, float bv, int bi) {
  return (av > bv) || (av == bv && ai < bi);
}
static __device__ __forceinline__ void top5_insert(float (&tv)[5], int (&ti)[5], float v, int id) {
  if (better(v, id, tv[4], ti[4])) {
    tv[4] = v; ti[4] = id;
#pragma unroll
    for (int j = 4; j > 0; --j) {
      if (better(tv[j], ti[j], tv[j - 1], ti[j - 1])) {
        float fv = tv[j]; tv[j] = tv[j - 1]; tv[j - 1] = fv;
        int   fi = ti[j]; ti[j] = ti[j - 1]; ti[j - 1] = fi;
      }
    }
  }
}

// Pack 16 fp32 -> bf16 fragment (element order = K 0..7,16..23 per 16-bit A layout).
static __device__ __forceinline__ v16bf packA(const float4 (&f)[4]) {
  v16bf r;
#pragma unroll
  for (int i = 0; i < 4; ++i) {
    r[4 * i + 0] = (__bf16)f[i].x;
    r[4 * i + 1] = (__bf16)f[i].y;
    r[4 * i + 2] = (__bf16)f[i].z;
    r[4 * i + 3] = (__bf16)f[i].w;
  }
  return r;
}

// ---------------- Kernel 0: q = query @ W^T + b  -> bf16 [64][768] ----------------
__global__ void __launch_bounds__(256)
embed_kernel(const float* __restrict__ query, const float* __restrict__ W,
             const float* __restrict__ bias, unsigned short* __restrict__ q_bf) {
  int idx = blockIdx.x * 256 + threadIdx.x;     // [0, 64*768)
  int n  = idx / DIMS;
  int kd = idx - n * DIMS;
  const float* qr = query + n * DIMS;
  const float* wr = W + (size_t)kd * DIMS;
  float acc = bias[kd];
#pragma unroll 4
  for (int d = 0; d < DIMS; ++d) acc += qr[d] * wr[d];
  q_bf[idx] = __builtin_bit_cast(unsigned short, (__bf16)acc);
}

// ---------------- Kernel 1: scores via bf16 WMMA + per-tile exact top-5 ----------------
// A (block_emb, fp32->bf16) streams from HBM on LOADcnt; B (q) is block-resident in LDS
// on DScnt, so WMMA waits never drain the HBM prefetch pipeline.
__global__ void __launch_bounds__(256)
score_kernel(const float* __restrict__ block_emb, const unsigned short* __restrict__ q_bf,
             float* __restrict__ cand_val, int* __restrict__ cand_id) {
  __shared__ union {
    unsigned short q[NQ][QPITCH];               // 99,328 B (row stride 388 dwords, %64==4)
    float sc[NQ][TILE_DOCS + 4];                // 33,792 B (aliased; q is dead by then)
  } smem;

  const int tile   = blockIdx.x;
  const int t      = threadIdx.x;
  const int L      = t & 31;          // lane in wave32
  const int w      = t >> 5;          // wave id 0..7 -> 16 docs each
  const int lane16 = L & 15;
  const int hi     = (L >> 4) & 1;

  // Cooperative copy q (64 x 768 bf16 = 6144 uint4) into padded LDS.
  {
    const uint4* src = (const uint4*)q_bf;
    for (int i = t; i < NQ * 96; i += 256) {
      const int row = i / 96;
      const int col = i - row * 96;
      ((uint4*)&smem.q[row][0])[col] = src[row * 96 + col];
    }
  }
  __syncthreads();

  // A: lane = doc row (M = L%16); lanes 0-15 hold K 0-7 & 16-23, lanes 16-31 hold K 8-15 & 24-31
  const int arow = min(tile * TILE_DOCS + w * 16 + lane16, NDOCS - 1);
  const float* abase = block_emb + (size_t)arow * DIMS + hi * 8;

  // B: lane = query column (N = L%16); lanes 0-15 hold K 0-7 & 16-23, lanes 16-31 hold K 8-15 & 24-31
  const uint4* bbase[4];
#pragma unroll
  for (int nt = 0; nt < 4; ++nt)
    bbase[nt] = (const uint4*)(&smem.q[nt * 16 + lane16][hi * 8]);

  float4 fa[2][4];        // raw fp32 A, double buffered (global, LOADcnt)
  uint4  fb[2][4][2];     // raw bf16 B fragments, double buffered (LDS, DScnt)

  auto loadg = [&](int kc, int buf) {
    const float* ap = abase + kc * 32;
    fa[buf][0] = *(const float4*)(ap);
    fa[buf][1] = *(const float4*)(ap + 4);
    fa[buf][2] = *(const float4*)(ap + 16);
    fa[buf][3] = *(const float4*)(ap + 20);
    const int bo = kc * 4;            // kc*32 bf16 = 4 uint4
#pragma unroll
    for (int nt = 0; nt < 4; ++nt) {
      fb[buf][nt][0] = bbase[nt][bo];
      fb[buf][nt][1] = bbase[nt][bo + 2];
    }
  };

  v8f z{};
  v8f c[4]; c[0] = z; c[1] = z; c[2] = z; c[3] = z;

  loadg(0, 0);
#pragma unroll
  for (int kc = 0; kc < KCHUNKS; ++kc) {
    const int cur = kc & 1;
    if (kc + 1 < KCHUNKS) loadg(kc + 1, cur ^ 1);   // prefetch next chunk (overlaps WMMAs)
    v16bf a = packA(fa[cur]);
#pragma unroll
    for (int nt = 0; nt < 4; ++nt) {
      union { uint4 q[2]; v16bf v; } b;
      b.q[0] = fb[cur][nt][0];
      b.q[1] = fb[cur][nt][1];
      c[nt] = __builtin_amdgcn_wmma_f32_16x16x32_bf16(
          false, a, false, b.v, (short)0, c[nt], false, false);
    }
  }

  __syncthreads();   // all waves done reading smem.q before it is overwritten with scores

  // C/D layout: lane(0-15)->N=lane,M=vgpr; lane(16-31)->N=lane-16,M=8+vgpr
  const int dloc_base = w * 16 + hi * 8;
  if (tile == NTILES - 1) {           // uniform branch: tail masking only in last block
#pragma unroll
    for (int nt = 0; nt < 4; ++nt) {
      const int n = nt * 16 + lane16;
#pragma unroll
      for (int v = 0; v < 8; ++v) {
        const int dloc = dloc_base + v;
        float sc = c[nt][v];
        if (tile * TILE_DOCS + dloc >= NDOCS) sc = -__builtin_huge_valf();
        smem.sc[n][dloc] = sc;
      }
    }
  } else {
#pragma unroll
    for (int nt = 0; nt < 4; ++nt) {
      const int n = nt * 16 + lane16;
#pragma unroll
      for (int v = 0; v < 8; ++v)
        smem.sc[n][dloc_base + v] = c[nt][v];
    }
  }
  __syncthreads();

  if (t < NQ) {
    float tv[5]; int ti[5];
#pragma unroll
    for (int j = 0; j < 5; ++j) { tv[j] = -__builtin_huge_valf(); ti[j] = 0x7fffffff; }
    for (int d = 0; d < TILE_DOCS; ++d)
      top5_insert(tv, ti, smem.sc[t][d], tile * TILE_DOCS + d);
    float* cv = cand_val + ((size_t)tile * NQ + t) * TOPK;
    int*   ci = cand_id  + ((size_t)tile * NQ + t) * TOPK;
#pragma unroll
    for (int j = 0; j < TOPK; ++j) { cv[j] = tv[j]; ci[j] = ti[j]; }
  }
}

// ---------------- Kernel 2: merge per-tile candidates -> global top-5 per query ----------------
__global__ void __launch_bounds__(256)
merge_kernel(const float* __restrict__ cand_val, const int* __restrict__ cand_id,
             int* __restrict__ out) {
  __shared__ float s_v[256 * TOPK];
  __shared__ int   s_i[256 * TOPK];
  const int q = blockIdx.x;
  const int t = threadIdx.x;
  const int total = NTILES * TOPK;

  float tv[5]; int ti[5];
#pragma unroll
  for (int j = 0; j < 5; ++j) { tv[j] = -__builtin_huge_valf(); ti[j] = 0x7fffffff; }

  for (int e = t; e < total; e += 256) {
    const int tile = e / TOPK;
    const int j    = e - tile * TOPK;
    const size_t idx = ((size_t)tile * NQ + q) * TOPK + j;
    top5_insert(tv, ti, cand_val[idx], cand_id[idx]);
  }
#pragma unroll
  for (int j = 0; j < 5; ++j) { s_v[t * 5 + j] = tv[j]; s_i[t * 5 + j] = ti[j]; }
  __syncthreads();

  for (int s = 128; s > 0; s >>= 1) {
    if (t < s) {
      float av[5], bv[5], mv[5]; int ai[5], bi[5], mi[5];
#pragma unroll
      for (int j = 0; j < 5; ++j) {
        av[j] = s_v[t * 5 + j];       ai[j] = s_i[t * 5 + j];
        bv[j] = s_v[(t + s) * 5 + j]; bi[j] = s_i[(t + s) * 5 + j];
      }
      int i = 0, k = 0;
#pragma unroll
      for (int m = 0; m < 5; ++m) {   // both lists sorted desc; i,k <= m <= 4 when read
        const bool ta = better(av[i], ai[i], bv[k], bi[k]);
        mv[m] = ta ? av[i] : bv[k];
        mi[m] = ta ? ai[i] : bi[k];
        if (ta) ++i; else ++k;
      }
#pragma unroll
      for (int j = 0; j < 5; ++j) { s_v[t * 5 + j] = mv[j]; s_i[t * 5 + j] = mi[j]; }
    }
    __syncthreads();
  }
  if (t < TOPK) out[q * TOPK + t] = s_i[t];
}

extern "C" void kernel_launch(void* const* d_in, const int* in_sizes, int n_in,
                              void* d_out, int out_size, void* d_ws, size_t ws_size,
                              hipStream_t stream) {
  (void)in_sizes; (void)n_in; (void)out_size; (void)ws_size;
  const float* query     = (const float*)d_in[0];
  const float* W         = (const float*)d_in[1];
  const float* bias      = (const float*)d_in[2];
  const float* block_emb = (const float*)d_in[3];
  // d_in[4] = top_k (device scalar) — fixed at 5 per reference.

  char* base = (char*)d_ws;
  unsigned short* q_bf = (unsigned short*)base;                 // 64*768*2 = 96 KB
  const size_t cand_off   = 128 * 1024;
  const size_t cand_bytes = (size_t)NTILES * NQ * TOPK * 4;     // ~5 MB each
  float* cand_val = (float*)(base + cand_off);
  int*   cand_id  = (int*)  (base + cand_off + cand_bytes);
  int*   out      = (int*)d_out;

  embed_kernel<<<(NQ * DIMS) / 256, 256, 0, stream>>>(query, W, bias, q_bf);
  score_kernel<<<NTILES, 256, 0, stream>>>(block_emb, q_bf, cand_val, cand_id);
  merge_kernel<<<NQ, 256, 0, stream>>>(cand_val, cand_id, out);
}
